// TransformerBlock_29506425323993
// MI455X (gfx1250) — compile-verified, matrix operand order checked by reading, not run on hardware
//
#include <hip/hip_runtime.h>
#include <hip/hip_bf16.h>
#include <stdint.h>

// ---------------------------------------------------------------------------
// Transformer block for MI455X (gfx1250, wave32, WMMA).
// S=2048, D=1024, H=16, DH=64, FF=4096. fp32 in/out, f16 WMMA internally.
// GEMMs: double-buffered GLOBAL_LOAD_ASYNC_TO_LDS staging + wmma_f32_16x16x32_f16.
// ---------------------------------------------------------------------------

#define TS 2048   // sequence length S
#define TD 1024   // model dim D
#define TH 16     // heads
#define TDH 64    // head dim
#define TFF 4096  // mlp dim

typedef __attribute__((ext_vector_type(16))) _Float16 v16h;
typedef __attribute__((ext_vector_type(8)))  _Float16 v8h;
typedef __attribute__((ext_vector_type(8)))  float    v8f;

union V16U { v16h v; v8h h[2]; };

static __device__ inline v16h mk16(v8h lo, v8h hi) {
  V16U u; u.h[0] = lo; u.h[1] = hi; return u.v;
}

static __device__ inline float gelu_f(float x) {
  float x3 = x * x * x;
  return 0.5f * x * (1.0f + tanhf(0.7978845608028654f * (x + 0.044715f * x3)));
}

// --- CDNA5 async global->LDS copy (ASYNCcnt tracked) -----------------------
// LDS byte address comes from the vdst VGPR; per-lane 16B transfer, no VGPR data.
static __device__ inline void async_ld_b128(uint32_t lds_off,
                                            const _Float16* gaddr) {
  asm volatile("global_load_async_to_lds_b128 %0, %1, off"
               :: "v"(lds_off), "v"(gaddr)
               : "memory");
}
static __device__ inline void wait_async_le4() {
  asm volatile("s_wait_asynccnt 0x4" ::: "memory");
}
static __device__ inline void wait_async_0() {
  asm volatile("s_wait_asynccnt 0x0" ::: "memory");
}

// ---------------------------------------------------------------------------
// LayerNorm (f32 in) -> f16 out. One block (256 thr) per row of D=1024.
// ---------------------------------------------------------------------------
__global__ __launch_bounds__(256) void layernorm_f16(
    const float* __restrict__ x, const float* __restrict__ w,
    const float* __restrict__ b, _Float16* __restrict__ y, int D) {
  const int row = blockIdx.x;
  const int t = threadIdx.x;
  const int lane = t & 31, wv = t >> 5;
  const float4 xv = ((const float4*)(x + (size_t)row * D))[t];
  float s = xv.x + xv.y + xv.z + xv.w;
  float ss = xv.x * xv.x + xv.y * xv.y + xv.z * xv.z + xv.w * xv.w;
#pragma unroll
  for (int off = 16; off >= 1; off >>= 1) {
    s += __shfl_xor(s, off, 32);
    ss += __shfl_xor(ss, off, 32);
  }
  __shared__ float rs[8], rss[8], mu_s, rstd_s;
  if (lane == 0) { rs[wv] = s; rss[wv] = ss; }
  __syncthreads();
  if (t == 0) {
    float S1 = 0.f, S2 = 0.f;
#pragma unroll
    for (int i = 0; i < 8; ++i) { S1 += rs[i]; S2 += rss[i]; }
    float mu = S1 / (float)D;
    float var = S2 / (float)D - mu * mu;
    mu_s = mu;
    rstd_s = rsqrtf(var + 1e-5f);
  }
  __syncthreads();
  const float mu = mu_s, rstd = rstd_s;
  const int c0 = t * 4;
  _Float16* yr = y + (size_t)row * D;
  float xa[4] = {xv.x, xv.y, xv.z, xv.w};
#pragma unroll
  for (int i = 0; i < 4; ++i) {
    int c = c0 + i;
    yr[c] = (_Float16)((xa[i] - mu) * rstd * w[c] + b[c]);
  }
}

// ---------------------------------------------------------------------------
// Weight transpose + f32->f16 convert: Wt[n*K + k] = (f16) W[k*N + n]
// ---------------------------------------------------------------------------
__global__ __launch_bounds__(256) void convertT_f16(
    const float* __restrict__ W, _Float16* __restrict__ Wt, int K, int N) {
  size_t idx = (size_t)blockIdx.x * 256 + threadIdx.x;
  size_t total = (size_t)K * N;
  if (idx >= total) return;
  int n = (int)(idx / K);
  int k = (int)(idx % K);
  Wt[idx] = (_Float16)W[(size_t)k * N + n];
}

// ---------------------------------------------------------------------------
// qkv (f16, row-major [S, 3*D]) -> Q [H][S][DH], K [H][S][DH], Vt [H][DH][S]
// ---------------------------------------------------------------------------
__global__ __launch_bounds__(256) void reshape_qkv(
    const _Float16* __restrict__ qkv, _Float16* __restrict__ Q,
    _Float16* __restrict__ Kh, _Float16* __restrict__ Vt) {
  size_t idx = (size_t)blockIdx.x * 256 + threadIdx.x;
  if (idx >= (size_t)TS * 3 * TD) return;
  int s = (int)(idx / (3 * TD));
  int n = (int)(idx % (3 * TD));
  int part = n / TD;
  int r = n - part * TD;
  int h = r >> 6;       // /64
  int d = r & 63;
  _Float16 v = qkv[idx];
  if (part == 0)      Q[((size_t)h * TS + s) * TDH + d] = v;
  else if (part == 1) Kh[((size_t)h * TS + s) * TDH + d] = v;
  else                Vt[((size_t)h * TDH + d) * TS + s] = v;
}

// ---------------------------------------------------------------------------
// Tiled WMMA GEMM: C[M,N] = A[M,K](f16,row-major) * Bt[N,K](f16, B transposed)
// BM=128, BN=128, BK=32. 256 threads = 8 waves (4 in M x 2 in N); each wave
// computes a 32x64 tile = 8 x v_wmma_f32_16x16x32_f16 per K-step.
// Double-buffered async global->LDS staging (4 async b128 per thread/tile).
// MODE: 0 = f16 store; 1 = +resid -> f32; 2 = +bias,gelu -> f16;
//       3 = +bias,+resid -> f32
// ---------------------------------------------------------------------------
#define GBM 128
#define GBN 128
#define GBK 32
#define LSTRIDE 40  // halves per LDS row (pad vs 32 to dodge bank conflicts)

template <int MODE>
__global__ __launch_bounds__(256) void gemm_wmma_f16(
    const _Float16* __restrict__ A, const _Float16* __restrict__ Bt,
    int M, int N, int K,
    const float* __restrict__ bias, const float* __restrict__ resid,
    _Float16* __restrict__ out_h, float* __restrict__ out_f) {
  __shared__ __align__(16) _Float16 As[2][GBM * LSTRIDE];
  __shared__ __align__(16) _Float16 Bs[2][GBN * LSTRIDE];
  const int t = threadIdx.x;
  const int lane = t & 31;
  const int wv = t >> 5;
  const int wm = wv >> 1;   // 0..3 -> 32-row band
  const int wn = wv & 1;    // 0..1 -> 64-col band
  const int l16 = lane & 15;
  const int sel = lane >> 4;
  const int n0 = blockIdx.x * GBN;
  const int m0 = blockIdx.y * GBM;

  // Per-thread staging coordinates (2 x b128 for A, 2 x b128 for B per tile)
  const int srow = t >> 2;          // 0..63
  const int sseg = (t & 3) * 8;     // 0,8,16,24 halves
  const uint32_t asA = (uint32_t)(uintptr_t)&As[0][0];
  const uint32_t asB = (uint32_t)(uintptr_t)&Bs[0][0];

  v8f acc[2][4];
#pragma unroll
  for (int i = 0; i < 2; ++i)
#pragma unroll
    for (int j = 0; j < 4; ++j)
#pragma unroll
      for (int r = 0; r < 8; ++r) acc[i][j][r] = 0.0f;

  auto stage = [&](int buf, int k0) {
#pragma unroll
    for (int i = 0; i < 2; ++i) {
      int row = srow + i * 64;
      async_ld_b128(asA + (uint32_t)(buf * GBM * LSTRIDE + row * LSTRIDE + sseg) * 2,
                    &A[(size_t)(m0 + row) * K + k0 + sseg]);
      async_ld_b128(asB + (uint32_t)(buf * GBN * LSTRIDE + row * LSTRIDE + sseg) * 2,
                    &Bt[(size_t)(n0 + row) * K + k0 + sseg]);
    }
  };

  const int ntiles = K / GBK;
  stage(0, 0);

  for (int kt = 0; kt < ntiles; ++kt) {
    const int cur = kt & 1;
    if (kt + 1 < ntiles) {
      stage(1 - cur, (kt + 1) * GBK);
      wait_async_le4();  // async loads retire in order: current tile done
    } else {
      wait_async_0();
    }
    __syncthreads();

    v16h a[2], b[4];
#pragma unroll
    for (int f = 0; f < 2; ++f) {
      const _Float16* ar = &As[cur][(wm * 32 + f * 16 + l16) * LSTRIDE];
      a[f] = mk16(*(const v8h*)&ar[sel * 8], *(const v8h*)&ar[16 + sel * 8]);
    }
#pragma unroll
    for (int f = 0; f < 4; ++f) {
      const _Float16* br = &Bs[cur][(wn * 64 + f * 16 + l16) * LSTRIDE];
      b[f] = mk16(*(const v8h*)&br[sel * 8], *(const v8h*)&br[16 + sel * 8]);
    }
#pragma unroll
    for (int fm = 0; fm < 2; ++fm)
#pragma unroll
      for (int fn = 0; fn < 4; ++fn)
        acc[fm][fn] = __builtin_amdgcn_wmma_f32_16x16x32_f16(
            false, a[fm], false, b[fn], (short)0, acc[fm][fn], false, false);
    __syncthreads();  // protect 'cur' buffer before it is refilled
  }

  // Epilogue. C layout: lane col = l16, row = r + 8*sel within 16x16 tile.
#pragma unroll
  for (int fm = 0; fm < 2; ++fm) {
#pragma unroll
    for (int fn = 0; fn < 4; ++fn) {
      const int nn = n0 + wn * 64 + fn * 16 + l16;
#pragma unroll
      for (int r = 0; r < 8; ++r) {
        const int mm = m0 + wm * 32 + fm * 16 + r + 8 * sel;
        float v = acc[fm][fn][r];
        if (MODE == 0) {
          out_h[(size_t)mm * N + nn] = (_Float16)v;
        } else if (MODE == 1) {
          out_f[(size_t)mm * N + nn] = v + resid[(size_t)mm * N + nn];
        } else if (MODE == 2) {
          out_h[(size_t)mm * N + nn] = (_Float16)gelu_f(v + bias[nn]);
        } else {
          out_f[(size_t)mm * N + nn] =
              v + bias[nn] + resid[(size_t)mm * N + nn];
        }
      }
    }
  }
}

// ---------------------------------------------------------------------------
// Flash attention, one wave per (head, 16-query tile). Computes S^T = K*Q^T
// so the softmax row lives in-lane (8 keys/lane + shfl_xor 16) and P^T is
// already in B-fragment half layout (k = e + 8*sel). O^T = V^T * P^T.
// ---------------------------------------------------------------------------
__global__ __launch_bounds__(256) void attention_wmma(
    const _Float16* __restrict__ Q, const _Float16* __restrict__ Kh,
    const _Float16* __restrict__ Vt, _Float16* __restrict__ attn) {
  const int t = threadIdx.x;
  const int lane = t & 31;
  const int wv = t >> 5;
  const int task = blockIdx.x * 8 + wv;  // H * S/16 = 2048 tasks
  const int h = task >> 7;               // /128
  const int qt = task & 127;
  const int q0 = qt * 16;
  const int l16 = lane & 15;
  const int sel = lane >> 4;

  const _Float16* Qb = Q + (size_t)h * TS * TDH;
  const _Float16* Kb = Kh + (size_t)h * TS * TDH;
  const _Float16* Vb = Vt + (size_t)h * TDH * TS;

  // Q fragments (B layout for S^T = K * Q^T): lane col = query
  v16h bq[2];
#pragma unroll
  for (int db = 0; db < 2; ++db) {
    const _Float16* qr = Qb + (size_t)(q0 + l16) * TDH + db * 32;
    bq[db] = mk16(*(const v8h*)&qr[sel * 8], *(const v8h*)&qr[16 + sel * 8]);
  }

  v8f o[4];
#pragma unroll
  for (int f = 0; f < 4; ++f)
#pragma unroll
    for (int r = 0; r < 8; ++r) o[f][r] = 0.0f;
  float mrow = -1e30f, lrow = 0.0f;
  const int q_abs = q0 + l16;

  for (int jt = 0; jt <= qt; ++jt) {
    const int j0 = jt * 16;
    // K tile as A fragment: lane row = key
    v16h ak[2];
#pragma unroll
    for (int db = 0; db < 2; ++db) {
      const _Float16* kr = Kb + (size_t)(j0 + l16) * TDH + db * 32;
      ak[db] = mk16(*(const v8h*)&kr[sel * 8], *(const v8h*)&kr[16 + sel * 8]);
    }
    v8f st;
#pragma unroll
    for (int r = 0; r < 8; ++r) st[r] = 0.0f;
    st = __builtin_amdgcn_wmma_f32_16x16x32_f16(false, ak[0], false, bq[0],
                                                (short)0, st, false, false);
    st = __builtin_amdgcn_wmma_f32_16x16x32_f16(false, ak[1], false, bq[1],
                                                (short)0, st, false, false);
    // scale + causal mask; st[r] = scores[key = j0+r+8*sel][q = q0+l16]
    float sc[8];
#pragma unroll
    for (int r = 0; r < 8; ++r) {
      int key = j0 + r + 8 * sel;
      float v = st[r] * 0.125f;  // 1/sqrt(64)
      sc[r] = (key > q_abs) ? -1e30f : v;
    }
    float tmax = sc[0];
#pragma unroll
    for (int r = 1; r < 8; ++r) tmax = fmaxf(tmax, sc[r]);
    tmax = fmaxf(tmax, __shfl_xor(tmax, 16, 32));
    const float mnew = fmaxf(mrow, tmax);
    const float alpha = __expf(mrow - mnew);
    float p[8], ps = 0.0f;
#pragma unroll
    for (int r = 0; r < 8; ++r) {
      p[r] = __expf(sc[r] - mnew);
      ps += p[r];
    }
    ps += __shfl_xor(ps, 16, 32);
    lrow = lrow * alpha + ps;
    mrow = mnew;
#pragma unroll
    for (int f = 0; f < 4; ++f)
#pragma unroll
      for (int r = 0; r < 8; ++r) o[f][r] *= alpha;

    // P^T as B fragment: elements 0..7 <-> k = e + 8*sel, upper half zero
    v16h bp;
#pragma unroll
    for (int e = 0; e < 16; ++e) bp[e] = (_Float16)0.0f;
#pragma unroll
    for (int e = 0; e < 8; ++e) bp[e] = (_Float16)p[e];

    // O^T += V^T * P^T over 4 d-tiles; A elements 0..7 = Vt[d][j0+8*sel+e]
#pragma unroll
    for (int f = 0; f < 4; ++f) {
      const _Float16* vr = Vb + (size_t)(f * 16 + l16) * TS + j0 + sel * 8;
      v8h lo = *(const v8h*)vr;
      v16h av;
#pragma unroll
      for (int e = 0; e < 16; ++e) av[e] = (_Float16)0.0f;
#pragma unroll
      for (int e = 0; e < 8; ++e) av[e] = lo[e];
      o[f] = __builtin_amdgcn_wmma_f32_16x16x32_f16(false, av, false, bp,
                                                    (short)0, o[f], false,
                                                    false);
    }
  }

  const float inv_l = 1.0f / lrow;
#pragma unroll
  for (int f = 0; f < 4; ++f) {
    v8h ov;
#pragma unroll
    for (int r = 0; r < 8; ++r) ov[r] = (_Float16)(o[f][r] * inv_l);
    // d = f*16 + 8*sel + r (r contiguous) at row q = q0 + l16
    _Float16* orow =
        attn + (size_t)q_abs * TD + h * TDH + f * 16 + 8 * sel;
    *(v8h*)orow = ov;
  }
}

// ---------------------------------------------------------------------------
// Host-side orchestration
// ---------------------------------------------------------------------------
extern "C" void kernel_launch(void* const* d_in, const int* in_sizes, int n_in,
                              void* d_out, int out_size, void* d_ws,
                              size_t ws_size, hipStream_t stream) {
  const float* x       = (const float*)d_in[0];
  const float* W_qkv   = (const float*)d_in[1];
  const float* W_o     = (const float*)d_in[2];
  const float* W1      = (const float*)d_in[3];
  const float* b1      = (const float*)d_in[4];
  const float* W2      = (const float*)d_in[5];
  const float* b2      = (const float*)d_in[6];
  const float* ln_a_w  = (const float*)d_in[7];
  const float* ln_a_b  = (const float*)d_in[8];
  const float* ln_m_w  = (const float*)d_in[9];
  const float* ln_m_b  = (const float*)d_in[10];
  float* out = (float*)d_out;

  // Workspace carving (256B aligned). ~72 MB total.
  char* p = (char*)d_ws;
  auto carve = [&](size_t bytes) {
    void* r = (void*)p;
    p += (bytes + 255) & ~(size_t)255;
    return r;
  };
  _Float16* xn1_h  = (_Float16*)carve((size_t)TS * TD * 2);        // 4 MB
  _Float16* Wqkv_t = (_Float16*)carve((size_t)3 * TD * TD * 2);    // 6 MB
  _Float16* Wo_t   = (_Float16*)carve((size_t)TD * TD * 2);        // 2 MB
  _Float16* W1_t   = (_Float16*)carve((size_t)TFF * TD * 2);       // 8 MB
  _Float16* W2_t   = (_Float16*)carve((size_t)TD * TFF * 2);       // 8 MB
  _Float16* big    = (_Float16*)carve((size_t)TS * TFF * 2);       // 16 MB (qkv_h then g_h)
  _Float16* Qh     = (_Float16*)carve((size_t)TH * TS * TDH * 2);  // 4 MB
  _Float16* Kh     = (_Float16*)carve((size_t)TH * TS * TDH * 2);  // 4 MB
  _Float16* Vt     = (_Float16*)carve((size_t)TH * TDH * TS * 2);  // 4 MB
  _Float16* attn_h = (_Float16*)carve((size_t)TS * TD * 2);        // 4 MB
  float*    x2     = (float*)carve((size_t)TS * TD * 4);           // 8 MB
  _Float16* h_h    = (_Float16*)carve((size_t)TS * TD * 2);        // 4 MB
  _Float16* qkv_h  = big;
  _Float16* g_h    = big;

  // 1) LN(x) -> f16
  layernorm_f16<<<TS, 256, 0, stream>>>(x, ln_a_w, ln_a_b, xn1_h, TD);

  // 2) Convert+transpose weights to f16 [N][K]
  convertT_f16<<<(3 * TD * TD) / 256, 256, 0, stream>>>(W_qkv, Wqkv_t, TD, 3 * TD);
  convertT_f16<<<(TD * TD) / 256, 256, 0, stream>>>(W_o, Wo_t, TD, TD);
  convertT_f16<<<(TD * TFF) / 256, 256, 0, stream>>>(W1, W1_t, TD, TFF);
  convertT_f16<<<(TFF * TD) / 256, 256, 0, stream>>>(W2, W2_t, TFF, TD);

  // 3) qkv = xn1 @ W_qkv  -> f16 [S, 3D]
  gemm_wmma_f16<0><<<dim3((3 * TD) / GBN, TS / GBM), 256, 0, stream>>>(
      xn1_h, Wqkv_t, TS, 3 * TD, TD, nullptr, nullptr, qkv_h, nullptr);

  // 4) reshape to Q/K/V^T per head
  reshape_qkv<<<(TS * 3 * TD) / 256, 256, 0, stream>>>(qkv_h, Qh, Kh, Vt);

  // 5) causal flash attention -> attn_h f16 [S, D]
  attention_wmma<<<(TH * (TS / 16)) / 8, 256, 0, stream>>>(Qh, Kh, Vt, attn_h);

  // 6) x2 = x + attn @ W_o   (f32)
  gemm_wmma_f16<1><<<dim3(TD / GBN, TS / GBM), 256, 0, stream>>>(
      attn_h, Wo_t, TS, TD, TD, nullptr, x, nullptr, x2);

  // 7) h = LN(x2) -> f16
  layernorm_f16<<<TS, 256, 0, stream>>>(x2, ln_m_w, ln_m_b, h_h, TD);

  // 8) g = gelu(h @ W1 + b1) -> f16 [S, FF]
  gemm_wmma_f16<2><<<dim3(TFF / GBN, TS / GBM), 256, 0, stream>>>(
      h_h, W1_t, TS, TFF, TD, b1, nullptr, g_h, nullptr);

  // 9) out = x2 + g @ W2 + b2 -> f32
  gemm_wmma_f16<3><<<dim3(TD / GBN, TS / GBM), 256, 0, stream>>>(
      g_h, W2_t, TS, TD, TFF, b2, x2, nullptr, out);
}
